// ProteinFeaturesLigand_42176578846966
// MI455X (gfx1250) — matrix operand order repaired and checked by reading, not compiled, not run on hardware
//
#include <hip/hip_runtime.h>
#include <hip/hip_bf16.h>
#include <math.h>

#define BB 2
#define LL 2048
#define TK 30
#define MM 16

typedef __attribute__((ext_vector_type(16))) __bf16 v16bf;
typedef __attribute__((ext_vector_type(8)))  float  v8f;
typedef __attribute__((ext_vector_type(4)))  unsigned int u32x4;
typedef __attribute__((ext_vector_type(8)))  unsigned int u32x8;

union Frag { v16bf bf; u32x4 u4[2]; };

// ---------- output offsets (floats) ----------
// V, E, E_idx, Y_nodes, Y_edges, Y_m
#define V_OFF    0LL
#define E_OFF    8388608LL
#define EIDX_OFF 24117248LL
#define YN_OFF   24240128LL
#define YE_OFF   32628736LL
#define YM_OFF   99737600LL

// ---------- workspace offsets (bytes) ----------
#define WS_A5    0
#define WS_RF    245760
#define WS_DN    393216
#define WS_EI    884736
#define WS_DB    1376256
#define WS_WET   1867776
#define WS_WNT   1974272
#define WS_WYET  2015232

__constant__ int c_AI[24] = {0,2,3,4,1,1,1,1,0,0,0,4,4,3,0,2,3,4,2,3,4,2,3,2};
__constant__ int c_BI[24] = {0,2,3,4,0,2,3,4,2,3,4,2,3,2,1,1,1,1,0,0,0,4,4,3};
__constant__ int c_GRP[] = {0, 1, 18, 1, 2, 13, 14, 15, 16, 17, 18, 1, 2, 13, 14, 15, 16, 17, 18, 1, 2, 3, 4, 5, 6, 7, 8, 9, 10, 11, 12, 13, 14, 15, 16, 17, 18, 1, 2, 3, 4, 5, 6, 7, 8, 9, 10, 11, 12, 13, 14, 15, 16, 17, 18, 1, 2, 3, 3, 3, 3, 3, 3, 3, 3, 3, 3, 3, 3, 3, 3, 3, 4, 5, 6, 7, 8, 9, 10, 11, 12, 13, 14, 15, 16, 17, 18, 1, 2, 3, 3, 3, 3, 3, 3, 3, 3, 3, 3, 3, 3, 3, 3, 3, 4, 5, 6, 7, 8, 9, 10, 11, 12, 13, 14, 15, 16, 17, 18};
__constant__ int c_PER[] = {0, 1, 1, 2, 2, 2, 2, 2, 2, 2, 2, 3, 3, 3, 3, 3, 3, 3, 3, 4, 4, 4, 4, 4, 4, 4, 4, 4, 4, 4, 4, 4, 4, 4, 4, 4, 4, 5, 5, 5, 5, 5, 5, 5, 5, 5, 5, 5, 5, 5, 5, 5, 5, 5, 5, 6, 6, 6, 6, 6, 6, 6, 6, 6, 6, 6, 6, 6, 6, 6, 6, 6, 6, 6, 6, 6, 6, 6, 6, 6, 6, 6, 6, 6, 6, 6, 6, 7, 7, 7, 7, 7, 7, 7, 7, 7, 7, 7, 7, 7, 7, 7, 7, 7, 7, 7, 7, 7, 7, 7, 7, 7, 7, 7, 7, 7, 7, 7, 7};

__device__ __forceinline__ unsigned short f2bf(float f) {
  unsigned int u = __float_as_uint(f);
  u += 0x7FFFu + ((u >> 16) & 1u);
  return (unsigned short)(u >> 16);
}

// rbf: mu = linspace(2,22,16) step 20/15, sigma = 1.25
__device__ __forceinline__ void rbf16_to(unsigned short* dst, float d) {
#pragma unroll
  for (int i = 0; i < 16; ++i) {
    float mu = 2.0f + 1.3333333333f * (float)i;
    float t = (d - mu) * 0.8f;
    dst[i] = f2bf(__expf(-t * t));
  }
}

// ============================================================
// Shared 64-row x 128-col GEMM tile engine (bf16 WMMA, f32 acc)
// feat: shared [64][KPAD] bf16 (row-major K)
// WT:   global [128][KPAD] bf16 (column-of-output major -> contiguous K)
// Weight chunk staging via Tensor Data Mover:
//   2D tile, 128 rows x 64B, row stride KPAD*2 bytes, landing contiguous
//   in LDS at wchunk. Issued by wave 0 only (uniform SGPR descriptor),
//   completion via TENSORcnt, then workgroup barrier.
// ============================================================
template<int KPAD>
__device__ __forceinline__ void gemm64x128(
    const unsigned short* feat, unsigned short* wchunk,
    const unsigned short* WT, const float* bias, float* outBase, int tid)
{
  constexpr int KC = KPAD / 32;
  const int lane = tid & 31, wave = tid >> 5;
  const int half = lane >> 4, mcol = lane & 15;
  const unsigned ldsoff = (unsigned)(size_t)(void*)wchunk;       // wave-relative LDS byte addr
  const unsigned long long gbase = (unsigned long long)(size_t)WT;
  v8f acc[8] = {};
  for (int kc = 0; kc < KC; ++kc) {
    __syncthreads();                 // protect previous iteration's fragment reads
    if (tid < 32) {                  // wave 0, EXEC all-ones: issue TDM copy
      unsigned long long ga = gbase + (unsigned)(kc * 64);
      u32x4 g0;
      g0[0] = 1u;                                            // count=1 (valid D#)
      g0[1] = ldsoff;                                        // lds_addr
      g0[2] = (unsigned)ga;                                  // global_addr[31:0]
      g0[3] = (unsigned)((ga >> 32) & 0x01FFFFFFu) | 0x80000000u; // ga[56:32] | type=2
      u32x8 g1;
      g1[0] = 0x00030000u;                                   // data_size=3 (8B), no mask
      g1[1] = 0xFFFF0000u;                                   // tensor_dim0 = 0x7FFFFFFF (lo16)
      g1[2] = 0xFFFF7FFFu;                                   // td0 hi16 | tensor_dim1 lo16
      g1[3] = 0x00087FFFu;                                   // td1 hi16 | tile_dim0 = 8 (64B/row)
      g1[4] = 128u;                                          // tile_dim1 = 128 rows, tile_dim2=0
      g1[5] = (unsigned)(KPAD / 4);                          // tensor_dim0_stride (8B units)
      g1[6] = 0u;
      g1[7] = 0u;
      asm volatile("tensor_load_to_lds %0, %1" :: "s"(g0), "s"(g1) : "memory");
      __builtin_amdgcn_s_wait_tensorcnt(0);
    }
    __syncthreads();                 // LDS tile visible to all waves
    // A fragment: lane m = mcol, K packed per ISA 16-bit A layout
    Frag a;
    const unsigned short* fr = feat + (wave * 16 + mcol) * KPAD + kc * 32 + half * 8;
    a.u4[0] = *(const u32x4*)fr;
    a.u4[1] = *(const u32x4*)(fr + 16);
#pragma unroll
    for (int n = 0; n < 8; ++n) {
      Frag b;
      const unsigned short* wp = wchunk + (n * 16 + mcol) * 32 + half * 8;
      b.u4[0] = *(const u32x4*)wp;
      b.u4[1] = *(const u32x4*)(wp + 16);
      acc[n] = __builtin_amdgcn_wmma_f32_16x16x32_bf16(
          false, a.bf, false, b.bf, (short)0, acc[n], false, false);
    }
  }
  // epilogue: C/D layout: VGPR r -> (M = r + 8*half, N = mcol)
#pragma unroll
  for (int n = 0; n < 8; ++n) {
    int col = n * 16 + mcol;
    float bv = bias[col];
#pragma unroll
    for (int rr = 0; rr < 8; ++rr) {
      int row = wave * 16 + rr + 8 * half;
      outBase[(size_t)row * 128 + col] = acc[n][rr] + bv;
    }
  }
}

// ============================================================
// Kernel 1: convert/transpose weights to bf16 (zero-padded K)
// ============================================================
__global__ void prep_weights_kernel(const float* We, const float* Wn, const float* Wye,
                                    unsigned short* WeT, unsigned short* WnT,
                                    unsigned short* WyeT)
{
  int gid = blockIdx.x * 256 + threadIdx.x;
  if (gid < 128 * 416) {
    int n = gid / 416, k = gid % 416;
    WeT[gid] = f2bf(We[k * 128 + n]);
  } else if (gid < 128 * 416 + 128 * 160) {
    int t = gid - 128 * 416; int n = t / 160, k = t % 160;
    WnT[t] = (k < 148) ? f2bf(Wn[k * 128 + n]) : (unsigned short)0;
  } else if (gid < 128 * 416 + 128 * 160 + 128 * 32) {
    int t = gid - 128 * 416 - 128 * 160; int n = t / 32, k = t % 32;
    WyeT[t] = (k < 16) ? f2bf(Wye[k * 128 + n]) : (unsigned short)0;
  }
}

// ============================================================
// Kernel 2: geometry: A5 = [N,Ca,C,O,Cb], local frame (e1,e2,e3)
// ============================================================
__global__ void geom_kernel(const float* X, float* A5, float* Rfr)
{
  int idx = blockIdx.x * 256 + threadIdx.x;
  if (idx >= BB * LL) return;
  const float* x = X + (size_t)idx * 12;
  float Nx=x[0],Ny=x[1],Nz=x[2], Cax=x[3],Cay=x[4],Caz=x[5];
  float Cx=x[6],Cy=x[7],Cz=x[8], Ox=x[9],Oy=x[10],Oz=x[11];
  float bx=Cax-Nx, by=Cay-Ny, bz=Caz-Nz;
  float cx=Cx-Cax, cy=Cy-Cay, cz=Cz-Caz;
  float ax=by*cz-bz*cy, ay=bz*cx-bx*cz, az=bx*cy-by*cx;
  float Cbx=-0.58273431f*ax+0.56802827f*bx-0.54067466f*cx+Cax;
  float Cby=-0.58273431f*ay+0.56802827f*by-0.54067466f*cy+Cay;
  float Cbz=-0.58273431f*az+0.56802827f*bz-0.54067466f*cz+Caz;
  float* a5 = A5 + (size_t)idx * 15;
  a5[0]=Nx; a5[1]=Ny; a5[2]=Nz; a5[3]=Cax; a5[4]=Cay; a5[5]=Caz;
  a5[6]=Cx; a5[7]=Cy; a5[8]=Cz; a5[9]=Ox; a5[10]=Oy; a5[11]=Oz;
  a5[12]=Cbx; a5[13]=Cby; a5[14]=Cbz;
  // frame: v1 = N-Ca = -b ; v2 = C-Ca = c
  float v1x=-bx, v1y=-by, v1z=-bz;
  float inv1 = 1.0f / sqrtf(v1x*v1x+v1y*v1y+v1z*v1z);
  float e1x=v1x*inv1, e1y=v1y*inv1, e1z=v1z*inv1;
  float dot = e1x*cx + e1y*cy + e1z*cz;
  float u2x=cx-e1x*dot, u2y=cy-e1y*dot, u2z=cz-e1z*dot;
  float inv2 = 1.0f / sqrtf(u2x*u2x+u2y*u2y+u2z*u2z);
  float e2x=u2x*inv2, e2y=u2y*inv2, e2z=u2z*inv2;
  float e3x=e1y*e2z-e1z*e2y, e3y=e1z*e2x-e1x*e2z, e3z=e1x*e2y-e1y*e2x;
  float* R = Rfr + (size_t)idx * 9;
  R[0]=e1x; R[1]=e1y; R[2]=e1z; R[3]=e2x; R[4]=e2y; R[5]=e2z;
  R[6]=e3x; R[7]=e3y; R[8]=e3z;
}

// ============================================================
// Kernel 3: masked pairwise distances + iterative top-30 argmin
// one block (256 thr) per (b,l)
// ============================================================
__global__ __launch_bounds__(256) void topk_kernel(
    const float* X, const float* mask, const int* Ridx, const int* chain,
    float* Dnb, int* Eidx, int* Dbkt, float* outEidxF)
{
  __shared__ float sD[LL];
  __shared__ float sRed[256];
  __shared__ int   sRedI[256];
  int tid = threadIdx.x;
  int bid = blockIdx.x;
  int b = bid >> 11;
  const float* ca = X + ((size_t)bid * 4 + 1) * 3;
  float cax = ca[0], cay = ca[1], caz = ca[2];
  float ml = mask[bid];
  float lmax = -1e30f;
  for (int j = tid; j < LL; j += 256) {
    const float* cj = X + (((size_t)(b * LL + j)) * 4 + 1) * 3;
    float dx = cax - cj[0], dy = cay - cj[1], dz = caz - cj[2];
    float m2 = ml * mask[b * LL + j];
    float D = m2 * sqrtf(dx * dx + dy * dy + dz * dz + 1e-6f);
    sD[j] = D;
    lmax = fmaxf(lmax, D);
  }
  sRed[tid] = lmax;
  __syncthreads();
  for (int s = 128; s > 0; s >>= 1) {
    if (tid < s) sRed[tid] = fmaxf(sRed[tid], sRed[tid + s]);
    __syncthreads();
  }
  float Dmax = sRed[0];
  __syncthreads();
  for (int j = tid; j < LL; j += 256) {
    float m2 = ml * mask[b * LL + j];
    sD[j] = sD[j] + (1.0f - m2) * Dmax;
  }
  __syncthreads();
  for (int kk = 0; kk < TK; ++kk) {
    float bv = 3.3e38f; int bi = 1 << 30;
    for (int j = tid; j < LL; j += 256) {
      float v = sD[j];
      if (v < bv || (v == bv && j < bi)) { bv = v; bi = j; }
    }
    sRed[tid] = bv; sRedI[tid] = bi;
    __syncthreads();
    for (int s = 128; s > 0; s >>= 1) {
      if (tid < s) {
        float ov = sRed[tid + s]; int oi = sRedI[tid + s];
        if (ov < sRed[tid] || (ov == sRed[tid] && oi < sRedI[tid])) {
          sRed[tid] = ov; sRedI[tid] = oi;
        }
      }
      __syncthreads();
    }
    if (tid == 0) {
      int j = sRedI[0];
      int er = bid * TK + kk;
      Eidx[er] = j;
      Dnb[er]  = sRed[0];
      outEidxF[er] = (float)j;
      int off = Ridx[bid] - Ridx[b * LL + j];
      int ech = (chain[bid] == chain[b * LL + j]) ? 1 : 0;
      int dd = off + 32; dd = dd < 0 ? 0 : (dd > 64 ? 64 : dd);
      Dbkt[er] = ech ? dd : 65;
      sD[j] = 3.4e38f;
    }
    __syncthreads();
  }
}

// ============================================================
// Kernel 4: E = [Epos | rbf(Dnb) | rbf(24 pair dists)] @ We + be
// 64 edges x 128 per block, K = 416 = 13 * 32
// ============================================================
__global__ __launch_bounds__(128) void e_gemm_kernel(
    const float* Wp, const float* bp, const unsigned short* WeT, const float* be,
    const float* A5, const float* Dnb, const int* Eidx, const int* Dbkt, float* out)
{
  __shared__ unsigned short feat[64 * 416];
  __shared__ unsigned short wchunk[128 * 32];
  int tid = threadIdx.x;
  long long row0 = (long long)blockIdx.x * 64;
  int rl = tid >> 1, part = tid & 1;
  long long r = row0 + rl;
  int resi = (int)(r / TK);
  int b = resi >> 11;
  int j = Eidx[r];
  int resj = (b << 11) | j;
  unsigned short* f = feat + rl * 416;
  if (part == 0) {
    int d = Dbkt[r];
#pragma unroll
    for (int c = 0; c < 16; ++c) f[c] = f2bf(Wp[d * 16 + c] + bp[c]);
    rbf16_to(f + 16, Dnb[r]);
    for (int p = 1; p < 12; ++p) {
      int q = p - 1;
      const float* pa = A5 + (size_t)resi * 15 + c_AI[q] * 3;
      const float* pb = A5 + (size_t)resj * 15 + c_BI[q] * 3;
      float dx = pa[0]-pb[0], dy = pa[1]-pb[1], dz = pa[2]-pb[2];
      rbf16_to(f + 16 + p * 16, sqrtf(dx*dx + dy*dy + dz*dz + 1e-6f));
    }
  } else {
    for (int p = 12; p < 25; ++p) {
      int q = p - 1;
      const float* pa = A5 + (size_t)resi * 15 + c_AI[q] * 3;
      const float* pb = A5 + (size_t)resj * 15 + c_BI[q] * 3;
      float dx = pa[0]-pb[0], dy = pa[1]-pb[1], dz = pa[2]-pb[2];
      rbf16_to(f + 16 + p * 16, sqrtf(dx*dx + dy*dy + dz*dz + 1e-6f));
    }
  }
  gemm64x128<416>(feat, wchunk, WeT, be, out + (size_t)row0 * 128, tid);
}

// ============================================================
// Kernel 5: V = [rbf_AY(80) | Yt_lin(64) | f_ang(4) | pad] @ Wn + bn
// K = 160 (148 padded); 64 rows per block
// ============================================================
__global__ __launch_bounds__(128) void v_gemm_kernel(
    const float* Y, const int* Y_t, const float* Wt, const float* bt,
    const float* A5, const float* Rfr, const unsigned short* WnT, const float* bn,
    float* out)
{
  __shared__ unsigned short feat[64 * 160];
  __shared__ unsigned short wchunk[128 * 32];
  int tid = threadIdx.x;
  long long row0 = (long long)blockIdx.x * 64;
  int rl = tid >> 1, part = tid & 1;
  long long r = row0 + rl;
  int m = (int)(r & 15);
  int idx = (int)(r >> 4);
  const float* yp = Y + ((size_t)idx * 16 + m) * 3;
  float yx = yp[0], yy = yp[1], yz = yp[2];
  unsigned short* f = feat + rl * 160;
  if (part == 0) {
    const float* a5 = A5 + (size_t)idx * 15;
    for (int a = 0; a < 5; ++a) {
      float dx = a5[a*3]-yx, dy = a5[a*3+1]-yy, dz = a5[a*3+2]-yz;
      rbf16_to(f + a * 16, sqrtf(dx*dx + dy*dy + dz*dz + 1e-6f));
    }
    const float* R = Rfr + (size_t)idx * 9;
    float dx = yx - a5[3], dy = yy - a5[4], dz = yz - a5[5];
    float lv0 = R[0]*dx + R[1]*dy + R[2]*dz;
    float lv1 = R[3]*dx + R[4]*dy + R[5]*dz;
    float lv2 = R[6]*dx + R[7]*dy + R[8]*dz;
    float rxy = sqrtf(lv0*lv0 + lv1*lv1 + 1e-8f);
    float rxyz = sqrtf(lv0*lv0 + lv1*lv1 + lv2*lv2) + 1e-8f;
    f[144] = f2bf(lv0 / rxy);
    f[145] = f2bf(lv1 / rxy);
    f[146] = f2bf(rxy / rxyz);
    f[147] = f2bf(lv2 / rxyz);
    for (int c = 148; c < 160; ++c) f[c] = 0;
  } else {
    int yt = Y_t[(size_t)idx * 16 + m];
    int g = c_GRP[yt], pp = c_PER[yt];
    const float* w0 = Wt + (size_t)yt * 64;
    const float* w1 = Wt + (size_t)(120 + g) * 64;
    const float* w2 = Wt + (size_t)(139 + pp) * 64;
    for (int c = 0; c < 64; ++c) f[80 + c] = f2bf(w0[c] + w1[c] + w2[c] + bt[c]);
  }
  gemm64x128<160>(feat, wchunk, WnT, bn, out + (size_t)row0 * 128, tid);
}

// ============================================================
// Kernel 6: Y_nodes = Wyn[yt]+Wyn[120+g]+Wyn[139+p]+byn (one-hot gather),
//           plus Y_m passthrough
// ============================================================
__global__ __launch_bounds__(256) void ynodes_kernel(
    const int* Y_t, const float* Wyn, const float* byn, const float* Y_m,
    float* outYN, float* outYM)
{
  int gid = blockIdx.x * 256 + threadIdx.x;
  int row = gid >> 5;
  int c = (gid & 31) * 4;
  int yt = Y_t[row];
  int g = c_GRP[yt], pp = c_PER[yt];
  const float* w0 = Wyn + (size_t)yt * 128 + c;
  const float* w1 = Wyn + (size_t)(120 + g) * 128 + c;
  const float* w2 = Wyn + (size_t)(139 + pp) * 128 + c;
  float* o = outYN + (size_t)row * 128 + c;
#pragma unroll
  for (int i = 0; i < 4; ++i) o[i] = w0[i] + w1[i] + w2[i] + byn[c + i];
  if (c == 0) outYM[row] = Y_m[row];
}

// ============================================================
// Kernel 7: Y_edges = rbf(|Y_m - Y_m'|) @ Wye + bye ; K = 32 (16 padded)
// ============================================================
__global__ __launch_bounds__(128) void ye_gemm_kernel(
    const float* Y, const unsigned short* WyeT, const float* bye, float* out)
{
  __shared__ unsigned short feat[64 * 32];
  __shared__ unsigned short wchunk[128 * 32];
  int tid = threadIdx.x;
  long long row0 = (long long)blockIdx.x * 64;
  int rl = tid >> 1, part = tid & 1;
  long long r = row0 + rl;
  int mp = (int)(r & 15);
  int m = (int)((r >> 4) & 15);
  int idx = (int)(r >> 8);
  const float* yb = Y + (size_t)idx * 48;
  float dx = yb[m*3] - yb[mp*3], dy = yb[m*3+1] - yb[mp*3+1], dz = yb[m*3+2] - yb[mp*3+2];
  float d = sqrtf(dx*dx + dy*dy + dz*dz + 1e-6f);
  unsigned short* f = feat + rl * 32;
#pragma unroll
  for (int i = 0; i < 8; ++i) {
    int k = part * 8 + i;
    float mu = 2.0f + 1.3333333333f * (float)k;
    float t = (d - mu) * 0.8f;
    f[k] = f2bf(__expf(-t * t));
    f[16 + part * 8 + i] = 0;
  }
  gemm64x128<32>(feat, wchunk, WyeT, bye, out + (size_t)row0 * 128, tid);
}

// ============================================================
extern "C" void kernel_launch(void* const* d_in, const int* in_sizes, int n_in,
                              void* d_out, int out_size, void* d_ws, size_t ws_size,
                              hipStream_t stream)
{
  const float* X    = (const float*)d_in[0];
  const float* Y    = (const float*)d_in[1];
  const float* mask = (const float*)d_in[2];
  const float* Y_m  = (const float*)d_in[3];
  const int*   Ridx = (const int*)d_in[4];
  const int*   chn  = (const int*)d_in[5];
  const int*   Y_t  = (const int*)d_in[6];
  const float* Wp   = (const float*)d_in[7];
  const float* bp   = (const float*)d_in[8];
  const float* We   = (const float*)d_in[9];
  const float* be   = (const float*)d_in[10];
  const float* Wn   = (const float*)d_in[11];
  const float* bn   = (const float*)d_in[12];
  const float* Wt   = (const float*)d_in[13];
  const float* bt   = (const float*)d_in[14];
  const float* Wyn  = (const float*)d_in[15];
  const float* byn  = (const float*)d_in[16];
  const float* Wye  = (const float*)d_in[17];
  const float* bye  = (const float*)d_in[18];
  float* out = (float*)d_out;
  char* ws = (char*)d_ws;

  float* A5   = (float*)(ws + WS_A5);
  float* Rfr  = (float*)(ws + WS_RF);
  float* Dnb  = (float*)(ws + WS_DN);
  int*   Eidx = (int*)(ws + WS_EI);
  int*   Dbkt = (int*)(ws + WS_DB);
  unsigned short* WeT  = (unsigned short*)(ws + WS_WET);
  unsigned short* WnT  = (unsigned short*)(ws + WS_WNT);
  unsigned short* WyeT = (unsigned short*)(ws + WS_WYET);

  prep_weights_kernel<<<(77824 + 255) / 256, 256, 0, stream>>>(We, Wn, Wye, WeT, WnT, WyeT);
  geom_kernel<<<(BB * LL + 255) / 256, 256, 0, stream>>>(X, A5, Rfr);
  topk_kernel<<<BB * LL, 256, 0, stream>>>(X, mask, Ridx, chn, Dnb, Eidx, Dbkt,
                                           out + EIDX_OFF);
  e_gemm_kernel<<<(BB * LL * TK) / 64, 128, 0, stream>>>(Wp, bp, WeT, be, A5, Dnb,
                                                         Eidx, Dbkt, out + E_OFF);
  v_gemm_kernel<<<(BB * LL * MM) / 64, 128, 0, stream>>>(Y, Y_t, Wt, bt, A5, Rfr,
                                                         WnT, bn, out + V_OFF);
  ynodes_kernel<<<(BB * LL * MM * 32) / 256, 256, 0, stream>>>(Y_t, Wyn, byn, Y_m,
                                                               out + YN_OFF, out + YM_OFF);
  ye_gemm_kernel<<<(BB * LL * MM * MM) / 64, 128, 0, stream>>>(Y, WyeT, bye, out + YE_OFF);
}
